// pna4pretrain_73443940762083
// MI455X (gfx1250) — matrix operand order repaired and checked by reading, not compiled, not run on hardware
//
#include <hip/hip_runtime.h>

// ---------------------------------------------------------------------------
// Types for CDNA5 WMMA (gfx1250, wave32)
// ---------------------------------------------------------------------------
typedef __attribute__((ext_vector_type(16))) __bf16 v16bf;
typedef __attribute__((ext_vector_type(8)))  float  v8f;

struct __attribute__((aligned(16))) U4 { unsigned int x, y, z, w; };
struct U8 { U4 lo, hi; };

__device__ __forceinline__ v16bf frag_from(const unsigned short* plo,
                                           const unsigned short* phi) {
  U8 t;
  t.lo = *(const U4*)plo;
  t.hi = *(const U4*)phi;
  return __builtin_bit_cast(v16bf, t);
}

__device__ __forceinline__ unsigned short f2bf(float f) {
  union { float f; unsigned int u; } v; v.f = f;
  unsigned int u = v.u;
  u += 0x7FFFu + ((u >> 16) & 1u);   // round-to-nearest-even
  return (unsigned short)(u >> 16);
}

// order-preserving float<->int mapping for integer atomicMin/Max
__device__ __forceinline__ int ordf(float f) {
  union { float f; int i; } v; v.f = f;
  return v.i >= 0 ? v.i : (v.i ^ 0x7FFFFFFF);
}
__device__ __forceinline__ float unordf(int i) {
  union { float f; int i; } v;
  v.i = i >= 0 ? i : (i ^ 0x7FFFFFFF);
  return v.f;
}

#define WMMA_BF16(ACC, A, B) \
  __builtin_amdgcn_wmma_f32_16x16x32_bf16(false, (A), false, (B), (short)0, (ACC), false, false)

static constexpr int NN = 50000;
static constexpr int EE = 800000;
static constexpr int GG = 2048;

// ---------------------------------------------------------------------------
// Prep kernels: weight conversion to transposed bf16, combined edge weight
// ---------------------------------------------------------------------------
// dst[col][k] (bf16, [Ncols][Kpad]) = src[k][col] (f32 row-major [K][Ncols])
__global__ void pna_conv_transpose(const float* __restrict__ src,
                                   unsigned short* __restrict__ dst,
                                   int K, int Ncols, int Kpad) {
  int idx = blockIdx.x * 256 + threadIdx.x;
  if (idx >= Ncols * Kpad) return;
  int col = idx / Kpad, k = idx - col * Kpad;
  float v = (k < K) ? src[k * Ncols + col] : 0.0f;
  dst[idx] = f2bf(v);
}

// WmT[col][k], k<160:  k<128 -> Wpre[k][col]; 128<=k<144 -> (We@Wpre_c)[k-128][col]; else 0
__global__ void pna_build_wm(const float* __restrict__ Wpre,  // [192][64]
                             const float* __restrict__ We,    // [16][64]
                             unsigned short* __restrict__ WmT) {
  int idx = blockIdx.x * 256 + threadIdx.x;
  if (idx >= 64 * 160) return;
  int col = idx / 160, k = idx - col * 160;
  float v = 0.0f;
  if (k < 128) {
    v = Wpre[k * 64 + col];
  } else if (k < 144) {
    int r = k - 128;
    float s = 0.0f;
    for (int j = 0; j < 64; ++j) s += We[r * 64 + j] * Wpre[(128 + j) * 64 + col];
    v = s;
  }
  WmT[idx] = f2bf(v);
}

__global__ void pna_build_bm(const float* __restrict__ Wpre,
                             const float* __restrict__ bpre,
                             const float* __restrict__ be,
                             float* __restrict__ bm) {
  int c = threadIdx.x;
  if (c >= 64) return;
  float s = bpre[c];
  for (int j = 0; j < 16; ++j) s += be[j] * Wpre[(128 + j) * 64 + c];
  bm[c] = s;
}

// ---------------------------------------------------------------------------
// Degree / pooling-count / avg-log reduction / min-max init
// ---------------------------------------------------------------------------
__global__ void pna_deg(const int* __restrict__ dst, float* __restrict__ deg) {
  int e = blockIdx.x * 256 + threadIdx.x;
  if (e < EE) atomicAdd(&deg[dst[e]], 1.0f);
}

__global__ void pna_cnt(const int* __restrict__ batch, float* __restrict__ cnt) {
  int n = blockIdx.x * 256 + threadIdx.x;
  if (n < NN) atomicAdd(&cnt[batch[n]], 1.0f);
}

__global__ void pna_logsum(const float* __restrict__ deg, float* __restrict__ acc) {
  __shared__ float s[256];
  int i = blockIdx.x * 256 + threadIdx.x;
  s[threadIdx.x] = (i < NN) ? logf(deg[i] + 1.0f) : 0.0f;
  __syncthreads();
  for (int st = 128; st > 0; st >>= 1) {
    if (threadIdx.x < st) s[threadIdx.x] += s[threadIdx.x + st];
    __syncthreads();
  }
  if (threadIdx.x == 0) atomicAdd(acc, s[0]);
}

__global__ void pna_init_mnmx(int* __restrict__ mni, int* __restrict__ mxi) {
  int i = blockIdx.x * 256 + threadIdx.x;
  if (i < NN * 64) { mni[i] = 0x7FFFFFFF; mxi[i] = 0x80000000; }
}

// ---------------------------------------------------------------------------
// Edge kernel: WMMA message GEMM fused with atomic segment aggregation.
// Block = 128 threads (4 waves), tile = 16 edges, K = 160 (padded 144).
// ---------------------------------------------------------------------------
__global__ __launch_bounds__(128)
void pna_edge_msg(const float* __restrict__ h,          // [N][64]
                  const float* __restrict__ eattr,      // [E][16]
                  const int* __restrict__ src,
                  const int* __restrict__ dst,
                  const unsigned short* __restrict__ WmT, // [64][160] bf16
                  const float* __restrict__ bm,           // [64]
                  float* __restrict__ sum, float* __restrict__ sumsq,
                  int* __restrict__ mni, int* __restrict__ mxi) {
  __shared__ unsigned short As[16 * 160];
  __shared__ int dsts[16];
  __shared__ int srcs[16];
  const int e0 = blockIdx.x * 16;
  const int tid = threadIdx.x;
  if (tid < 16) { dsts[tid] = dst[e0 + tid]; srcs[tid] = src[e0 + tid]; }
  __syncthreads();
  for (int idx = tid; idx < 16 * 160; idx += 128) {
    int r = idx / 160, k = idx - r * 160;
    float v;
    if (k < 64)        v = h[dsts[r] * 64 + k];
    else if (k < 128)  v = h[srcs[r] * 64 + (k - 64)];
    else if (k < 144)  v = eattr[(e0 + r) * 16 + (k - 128)];
    else               v = 0.0f;
    As[idx] = f2bf(v);
  }
  __syncthreads();

  const int wid = tid >> 5, lane = tid & 31;
  const int col16 = lane & 15, half = lane >> 4;
  const int col = wid * 16 + col16;
  const unsigned short* arow = &As[col16 * 160];
  const unsigned short* brow = WmT + col * 160;
  v8f acc = {};
#pragma unroll
  for (int ks = 0; ks < 5; ++ks) {
    v16bf a = frag_from(arow + ks * 32 + half * 8, arow + ks * 32 + 16 + half * 8);
    v16bf b = frag_from(brow + ks * 32 + half * 8, brow + ks * 32 + 16 + half * 8);
    acc = WMMA_BF16(acc, a, b);
  }
  const float bias = bm[col];
#pragma unroll
  for (int i = 0; i < 8; ++i) {
    int r = i + 8 * half;
    float v = acc[i] + bias;
    int o = dsts[r] * 64 + col;
    atomicAdd(&sum[o], v);
    atomicAdd(&sumsq[o], v * v);
    atomicMin(&mni[o], ordf(v));
    atomicMax(&mxi[o], ordf(v));
  }
}

// ---------------------------------------------------------------------------
// Node kernel: finalize aggregations -> [16][832] bf16 A, WMMA x Wpost (K=832),
// bias, WMMA x Wlin (K=64), bias + relu -> h_out.
// ---------------------------------------------------------------------------
__global__ __launch_bounds__(128)
void pna_node_update(const float* __restrict__ h_in,
                     const float* __restrict__ sum, const float* __restrict__ sumsq,
                     const int* __restrict__ mni, const int* __restrict__ mxi,
                     const float* __restrict__ deg, const float* __restrict__ logacc,
                     const unsigned short* __restrict__ WpostT, // [64][832]
                     const float* __restrict__ bpost,
                     const unsigned short* __restrict__ WlinT,  // [64][64]
                     const float* __restrict__ blin,
                     float* __restrict__ h_out) {
  __shared__ unsigned short As[16 * 832];
  __shared__ unsigned short Ts[16 * 64];
  const int n0 = blockIdx.x * 16;
  const int tid = threadIdx.x;
  const float avg_log = logacc[0] / (float)NN;
  for (int idx = tid; idx < 16 * 64; idx += 128) {
    int t = idx >> 6, c = idx & 63;
    int n = n0 + t;
    int o = n * 64 + c;
    float dg = deg[n];
    float d1 = fmaxf(dg, 1.0f);
    float mean = sum[o] / d1;
    float m2   = sumsq[o] / d1;
    float stdv = sqrtf(fmaxf(m2 - mean * mean, 0.0f) + 1e-5f);
    bool has = dg > 0.0f;
    float mn = has ? unordf(mni[o]) : 0.0f;
    float mx = has ? unordf(mxi[o]) : 0.0f;
    float logd = logf(d1 + 1.0f);
    float amp = logd / avg_log;
    float att = avg_log / logd;
    unsigned short* row = &As[t * 832];
    row[c]       = f2bf(h_in[o]);
    row[64 + c]  = f2bf(mean);
    row[128 + c] = f2bf(mn);
    row[192 + c] = f2bf(mx);
    row[256 + c] = f2bf(stdv);
    row[320 + c] = f2bf(mean * amp);
    row[384 + c] = f2bf(mn * amp);
    row[448 + c] = f2bf(mx * amp);
    row[512 + c] = f2bf(stdv * amp);
    row[576 + c] = f2bf(mean * att);
    row[640 + c] = f2bf(mn * att);
    row[704 + c] = f2bf(mx * att);
    row[768 + c] = f2bf(stdv * att);
  }
  __syncthreads();

  const int wid = tid >> 5, lane = tid & 31;
  const int col16 = lane & 15, half = lane >> 4;
  const int col = wid * 16 + col16;
  {
    const unsigned short* arow = &As[col16 * 832];
    const unsigned short* brow = WpostT + col * 832;
    v8f acc = {};
#pragma unroll
    for (int ks = 0; ks < 26; ++ks) {
      v16bf a = frag_from(arow + ks * 32 + half * 8, arow + ks * 32 + 16 + half * 8);
      v16bf b = frag_from(brow + ks * 32 + half * 8, brow + ks * 32 + 16 + half * 8);
      acc = WMMA_BF16(acc, a, b);
    }
    float bb = bpost[col];
#pragma unroll
    for (int i = 0; i < 8; ++i)
      Ts[(i + 8 * half) * 64 + col] = f2bf(acc[i] + bb);
  }
  __syncthreads();
  {
    const unsigned short* arow = &Ts[col16 * 64];
    const unsigned short* brow = WlinT + col * 64;
    v8f acc = {};
#pragma unroll
    for (int ks = 0; ks < 2; ++ks) {
      v16bf a = frag_from(arow + ks * 32 + half * 8, arow + ks * 32 + 16 + half * 8);
      v16bf b = frag_from(brow + ks * 32 + half * 8, brow + ks * 32 + 16 + half * 8);
      acc = WMMA_BF16(acc, a, b);
    }
    float bl = blin[col];
#pragma unroll
    for (int i = 0; i < 8; ++i) {
      int n = n0 + i + 8 * half;
      h_out[n * 64 + col] = fmaxf(acc[i] + bl, 0.0f);
    }
  }
}

// ---------------------------------------------------------------------------
// JK concat GEMM (K=192) + atomic mean-pool accumulation into gsum
// ---------------------------------------------------------------------------
__global__ __launch_bounds__(128)
void pna_jk_pool(const float* __restrict__ h1, const float* __restrict__ h2,
                 const float* __restrict__ h3, const int* __restrict__ batch,
                 const unsigned short* __restrict__ WjkT, // [64][192]
                 const float* __restrict__ bjk,
                 float* __restrict__ gsum) {
  __shared__ unsigned short As[16 * 192];
  __shared__ int gidx[16];
  const int n0 = blockIdx.x * 16;
  const int tid = threadIdx.x;
  if (tid < 16) gidx[tid] = batch[n0 + tid];
  for (int idx = tid; idx < 16 * 192; idx += 128) {
    int t = idx / 192, k = idx - t * 192;
    int o = (n0 + t) * 64;
    float v = (k < 64) ? h1[o + k] : (k < 128 ? h2[o + k - 64] : h3[o + k - 128]);
    As[idx] = f2bf(v);
  }
  __syncthreads();

  const int wid = tid >> 5, lane = tid & 31;
  const int col16 = lane & 15, half = lane >> 4;
  const int col = wid * 16 + col16;
  const unsigned short* arow = &As[col16 * 192];
  const unsigned short* brow = WjkT + col * 192;
  v8f acc = {};
#pragma unroll
  for (int ks = 0; ks < 6; ++ks) {
    v16bf a = frag_from(arow + ks * 32 + half * 8, arow + ks * 32 + 16 + half * 8);
    v16bf b = frag_from(brow + ks * 32 + half * 8, brow + ks * 32 + 16 + half * 8);
    acc = WMMA_BF16(acc, a, b);
  }
  float bb = bjk[col];
#pragma unroll
  for (int i = 0; i < 8; ++i) {
    int r = i + 8 * half;
    atomicAdd(&gsum[gidx[r] * 64 + col], acc[i] + bb);
  }
}

// ---------------------------------------------------------------------------
// Head: g = gsum/cnt; out = relu(g@W1+b1)@W2+b2
// ---------------------------------------------------------------------------
__global__ __launch_bounds__(128)
void pna_head(const float* __restrict__ gsum, const float* __restrict__ cnt,
              const unsigned short* __restrict__ W1T, const float* __restrict__ b1,
              const unsigned short* __restrict__ W2T, const float* __restrict__ b2,
              float* __restrict__ out) {
  __shared__ unsigned short As[16 * 64];
  __shared__ unsigned short Ts[16 * 64];
  const int g0 = blockIdx.x * 16;
  const int tid = threadIdx.x;
  for (int idx = tid; idx < 16 * 64; idx += 128) {
    int t = idx >> 6, c = idx & 63;
    int g = g0 + t;
    float cn = fmaxf(cnt[g], 1.0f);
    As[idx] = f2bf(gsum[g * 64 + c] / cn);
  }
  __syncthreads();

  const int wid = tid >> 5, lane = tid & 31;
  const int col16 = lane & 15, half = lane >> 4;
  const int col = wid * 16 + col16;
  {
    const unsigned short* arow = &As[col16 * 64];
    const unsigned short* brow = W1T + col * 64;
    v8f acc = {};
#pragma unroll
    for (int ks = 0; ks < 2; ++ks) {
      v16bf a = frag_from(arow + ks * 32 + half * 8, arow + ks * 32 + 16 + half * 8);
      v16bf b = frag_from(brow + ks * 32 + half * 8, brow + ks * 32 + 16 + half * 8);
      acc = WMMA_BF16(acc, a, b);
    }
    float bb = b1[col];
#pragma unroll
    for (int i = 0; i < 8; ++i)
      Ts[(i + 8 * half) * 64 + col] = f2bf(fmaxf(acc[i] + bb, 0.0f));
  }
  __syncthreads();
  {
    const unsigned short* arow = &Ts[col16 * 64];
    const unsigned short* brow = W2T + col * 64;
    v8f acc = {};
#pragma unroll
    for (int ks = 0; ks < 2; ++ks) {
      v16bf a = frag_from(arow + ks * 32 + half * 8, arow + ks * 32 + 16 + half * 8);
      v16bf b = frag_from(brow + ks * 32 + half * 8, brow + ks * 32 + 16 + half * 8);
      acc = WMMA_BF16(acc, a, b);
    }
    float bb = b2[col];
#pragma unroll
    for (int i = 0; i < 8; ++i)
      out[(g0 + i + 8 * half) * 64 + col] = acc[i] + bb;
  }
}

// ---------------------------------------------------------------------------
// Host orchestration
// ---------------------------------------------------------------------------
extern "C" void kernel_launch(void* const* d_in, const int* in_sizes, int n_in,
                              void* d_out, int out_size, void* d_ws, size_t ws_size,
                              hipStream_t stream) {
  (void)in_sizes; (void)n_in; (void)out_size; (void)ws_size;
  const float* x     = (const float*)d_in[0];
  const int*   ei    = (const int*)d_in[1];
  const float* eattr = (const float*)d_in[2];
  const int*   batch = (const int*)d_in[3];
  const int* srcp = ei;
  const int* dstp = ei + EE;

  const float *We[3], *be[3], *Wpre[3], *bpre[3], *Wpost[3], *bpost[3], *Wlin[3], *blin[3];
  for (int l = 0; l < 3; ++l) {
    int b = 4 + l * 8;
    We[l]    = (const float*)d_in[b + 0];
    be[l]    = (const float*)d_in[b + 1];
    Wpre[l]  = (const float*)d_in[b + 2];
    bpre[l]  = (const float*)d_in[b + 3];
    Wpost[l] = (const float*)d_in[b + 4];
    bpost[l] = (const float*)d_in[b + 5];
    Wlin[l]  = (const float*)d_in[b + 6];
    blin[l]  = (const float*)d_in[b + 7];
  }
  const float* Wjk = (const float*)d_in[28];
  const float* bjk = (const float*)d_in[29];
  const float* W1  = (const float*)d_in[30];
  const float* b1  = (const float*)d_in[31];
  const float* W2  = (const float*)d_in[32];
  const float* b2  = (const float*)d_in[33];

  char* ws = (char*)d_ws;
  size_t off = 0;
  auto alloc = [&](size_t bytes) -> char* {
    char* p = ws + off;
    off += (bytes + 255) & ~(size_t)255;
    return p;
  };
  float* sum    = (float*)alloc((size_t)NN * 64 * 4);
  float* sumsq  = (float*)alloc((size_t)NN * 64 * 4);
  int*   mni    = (int*)  alloc((size_t)NN * 64 * 4);
  int*   mxi    = (int*)  alloc((size_t)NN * 64 * 4);
  float* h1     = (float*)alloc((size_t)NN * 64 * 4);
  float* h2     = (float*)alloc((size_t)NN * 64 * 4);
  float* h3     = (float*)alloc((size_t)NN * 64 * 4);
  float* deg    = (float*)alloc((size_t)NN * 4);
  float* logacc = (float*)alloc(256);
  float* gsum   = (float*)alloc((size_t)GG * 64 * 4);
  float* cnt    = (float*)alloc((size_t)GG * 4);
  unsigned short* WmT[3];
  float*          bm[3];
  unsigned short* WpostT[3];
  unsigned short* WlinT[3];
  for (int l = 0; l < 3; ++l) {
    WmT[l]    = (unsigned short*)alloc((size_t)64 * 160 * 2);
    bm[l]     = (float*)alloc(64 * 4);
    WpostT[l] = (unsigned short*)alloc((size_t)64 * 832 * 2);
    WlinT[l]  = (unsigned short*)alloc((size_t)64 * 64 * 2);
  }
  unsigned short* WjkT = (unsigned short*)alloc((size_t)64 * 192 * 2);
  unsigned short* W1T  = (unsigned short*)alloc((size_t)64 * 64 * 2);
  unsigned short* W2T  = (unsigned short*)alloc((size_t)64 * 64 * 2);

  // --- zero init ---
  hipMemsetAsync(deg,    0, (size_t)NN * 4, stream);
  hipMemsetAsync(logacc, 0, 4, stream);
  hipMemsetAsync(gsum,   0, (size_t)GG * 64 * 4, stream);
  hipMemsetAsync(cnt,    0, (size_t)GG * 4, stream);

  // --- weight prep ---
  for (int l = 0; l < 3; ++l) {
    pna_build_wm<<<(64 * 160 + 255) / 256, 256, 0, stream>>>(Wpre[l], We[l], WmT[l]);
    pna_build_bm<<<1, 64, 0, stream>>>(Wpre[l], bpre[l], be[l], bm[l]);
    pna_conv_transpose<<<(64 * 832 + 255) / 256, 256, 0, stream>>>(Wpost[l], WpostT[l], 832, 64, 832);
    pna_conv_transpose<<<(64 * 64 + 255) / 256, 256, 0, stream>>>(Wlin[l], WlinT[l], 64, 64, 64);
  }
  pna_conv_transpose<<<(64 * 192 + 255) / 256, 256, 0, stream>>>(Wjk, WjkT, 192, 64, 192);
  pna_conv_transpose<<<(64 * 64 + 255) / 256, 256, 0, stream>>>(W1, W1T, 64, 64, 64);
  pna_conv_transpose<<<(64 * 64 + 255) / 256, 256, 0, stream>>>(W2, W2T, 64, 64, 64);

  // --- degrees, counts, avg log ---
  pna_deg<<<(EE + 255) / 256, 256, 0, stream>>>(dstp, deg);
  pna_cnt<<<(NN + 255) / 256, 256, 0, stream>>>(batch, cnt);
  pna_logsum<<<(NN + 255) / 256, 256, 0, stream>>>(deg, logacc);

  // --- 3 PNA layers ---
  const float* h_in = x;
  float* h_out_arr[3] = {h1, h2, h3};
  for (int l = 0; l < 3; ++l) {
    hipMemsetAsync(sum,   0, (size_t)NN * 64 * 4, stream);
    hipMemsetAsync(sumsq, 0, (size_t)NN * 64 * 4, stream);
    pna_init_mnmx<<<(NN * 64 + 255) / 256, 256, 0, stream>>>(mni, mxi);

    pna_edge_msg<<<EE / 16, 128, 0, stream>>>(h_in, eattr, srcp, dstp,
                                              WmT[l], bm[l], sum, sumsq, mni, mxi);
    pna_node_update<<<NN / 16, 128, 0, stream>>>(h_in, sum, sumsq, mni, mxi, deg, logacc,
                                                 WpostT[l], bpost[l], WlinT[l], blin[l],
                                                 h_out_arr[l]);
    h_in = h_out_arr[l];
  }

  // --- JK + pooling + head ---
  pna_jk_pool<<<NN / 16, 128, 0, stream>>>(h1, h2, h3, batch, WjkT, bjk, gsum);
  pna_head<<<GG / 16, 128, 0, stream>>>(gsum, cnt, W1T, b1, W2T, b2, (float*)d_out);
}
